// NemotronHMoE_78374563218004
// MI455X (gfx1250) — compile-verified, matrix operand order checked by reading, not run on hardware
//
#include <hip/hip_runtime.h>
#include <math.h>
#include <stdint.h>

// ---- static config (matches reference) ----
#define HIDDEN        2048
#define MOE_INTER     1024
#define SHARED_INTER  4096
#define N_EXPERTS     16
#define T_TOK         2048
#define N_GROUP       4
#define GSZ           (N_EXPERTS / N_GROUP)

typedef __attribute__((ext_vector_type(16))) __bf16 v16bf;
typedef __attribute__((ext_vector_type(8)))  float  v8f;

// gfx1250 async global->LDS copy (ASYNCcnt-tracked), with sync fallback.
#if defined(__gfx1250__) && __has_builtin(__builtin_amdgcn_global_load_async_to_lds_b128)
  #define USE_ASYNC_LDS 1
#else
  #define USE_ASYNC_LDS 0
#endif

#if USE_ASYNC_LDS
typedef int v4i __attribute__((vector_size(16)));
typedef __attribute__((address_space(1))) v4i* gvp;   // global v4i*
typedef __attribute__((address_space(3))) v4i* lvp;   // LDS v4i*
static __device__ __forceinline__ void async_copy16(void* l, const void* g) {
  __builtin_amdgcn_global_load_async_to_lds_b128(
      (gvp)(unsigned long long)(uintptr_t)g,
      (lvp)(unsigned int)(uintptr_t)l, 0, 0);
}
static __device__ __forceinline__ void wait_async0() {
#if __has_builtin(__builtin_amdgcn_s_wait_asynccnt)
  __builtin_amdgcn_s_wait_asynccnt(0);
#else
  asm volatile("s_wait_asynccnt 0x0" ::: "memory");
#endif
}
#endif

static __device__ __forceinline__ unsigned short f2bf(float f) {
  union { float f; unsigned u; } v; v.f = f;
  unsigned r = v.u + 0x7FFFu + ((v.u >> 16) & 1u);   // round-to-nearest-even
  return (unsigned short)(r >> 16);
}
static __device__ __forceinline__ unsigned pack2bf(float a, float b) {
  return (unsigned)f2bf(a) | ((unsigned)f2bf(b) << 16);
}

// ------------------------------------------------------------------
// Router: one wave (32 lanes) per token; wave32 xor-shuffle reduction;
// lane0 does grouped top-k (first-index tie-break like jax.lax.top_k).
// ------------------------------------------------------------------
__global__ void __launch_bounds__(256) router_kernel(
    const float* __restrict__ x, const float* __restrict__ rw,
    const float* __restrict__ ebias,
    int* __restrict__ topk_idx, float* __restrict__ topk_w,
    int* __restrict__ counts) {
  int wid  = threadIdx.x >> 5;
  int lane = threadIdx.x & 31;
  int t = blockIdx.x * 8 + wid;
  if (t >= T_TOK) return;

  float acc[N_EXPERTS];
#pragma unroll
  for (int e = 0; e < N_EXPERTS; ++e) acc[e] = 0.f;
  const float* xr = x + (size_t)t * HIDDEN;
  for (int h = lane; h < HIDDEN; h += 32) {
    float xv = xr[h];
#pragma unroll
    for (int e = 0; e < N_EXPERTS; ++e) acc[e] += xv * rw[e * HIDDEN + h];
  }
#pragma unroll
  for (int e = 0; e < N_EXPERTS; ++e) {
#pragma unroll
    for (int off = 16; off > 0; off >>= 1)
      acc[e] += __shfl_xor(acc[e], off, 32);
  }
  if (lane != 0) return;

  float sc[N_EXPERTS], sb[N_EXPERTS];
#pragma unroll
  for (int e = 0; e < N_EXPERTS; ++e) {
    sc[e] = 1.f / (1.f + __expf(-acc[e]));
    sb[e] = sc[e] + ebias[e];
  }
  float gs[N_GROUP];
#pragma unroll
  for (int g = 0; g < N_GROUP; ++g) {
    float m1 = -1e30f, m2 = -1e30f;
#pragma unroll
    for (int j = 0; j < GSZ; ++j) {
      float v = sb[g * GSZ + j];
      if (v > m1) { m2 = m1; m1 = v; } else if (v > m2) { m2 = v; }
    }
    gs[g] = m1 + m2;
  }
  int g1 = 0;
#pragma unroll
  for (int g = 1; g < N_GROUP; ++g) if (gs[g] > gs[g1]) g1 = g;
  int g2 = -1;
#pragma unroll
  for (int g = 0; g < N_GROUP; ++g)
    if (g != g1 && (g2 < 0 || gs[g] > gs[g2])) g2 = g;

  int i1 = -1; float b1 = -1e30f;
#pragma unroll
  for (int e = 0; e < N_EXPERTS; ++e) {
    int g = e / GSZ;
    float v = (g == g1 || g == g2) ? sb[e] : 0.0f;
    if (v > b1) { b1 = v; i1 = e; }
  }
  int i2 = -1; float b2 = -1e30f;
#pragma unroll
  for (int e = 0; e < N_EXPERTS; ++e) {
    if (e == i1) continue;
    int g = e / GSZ;
    float v = (g == g1 || g == g2) ? sb[e] : 0.0f;
    if (v > b2) { b2 = v; i2 = e; }
  }
  float w1 = sc[i1], w2 = sc[i2];
  float s = w1 + w2 + 1e-20f;
  w1 /= s; w2 /= s;                 // NORM_TOPK_PROB, ROUTED_SCALING=1
  topk_idx[t * 2 + 0] = i1;  topk_w[t * 2 + 0] = w1;
  topk_idx[t * 2 + 1] = i2;  topk_w[t * 2 + 1] = w2;
  atomicAdd(&counts[i1], 1);
  atomicAdd(&counts[i2], 1);
}

__global__ void zero_counts_kernel(int* __restrict__ counts) {
  if (threadIdx.x < N_EXPERTS) counts[threadIdx.x] = 0;
}

__global__ void scan_kernel(const int* __restrict__ counts,
                            int* __restrict__ offsets,
                            int* __restrict__ cursors) {
  if (threadIdx.x == 0) {
    int run = 0;
    for (int e = 0; e < N_EXPERTS; ++e) {
      offsets[e] = run; cursors[e] = run; run += counts[e];
    }
    offsets[N_EXPERTS] = run;
  }
}

// one block per (token,k) pair: claim slot, record gate/token, copy x row (bf16)
__global__ void __launch_bounds__(256) gather_kernel(
    const float* __restrict__ x, const int* __restrict__ topk_idx,
    const float* __restrict__ topk_w, int* __restrict__ cursors,
    int* __restrict__ token_ids, float* __restrict__ gates,
    unsigned short* __restrict__ xg) {
  __shared__ int slot_s;
  int pair = blockIdx.x;
  int t = pair >> 1;
  if (threadIdx.x == 0) {
    int e = topk_idx[pair];
    int slot = atomicAdd(&cursors[e], 1);
    token_ids[slot] = t;
    gates[slot] = topk_w[pair];
    slot_s = slot;
  }
  __syncthreads();
  int slot = slot_s;
  const float* xr = x + (size_t)t * HIDDEN;
  unsigned short* dst = xg + (size_t)slot * HIDDEN;
  for (int h = threadIdx.x; h < HIDDEN; h += 256) dst[h] = f2bf(xr[h]);
}

// ------------------------------------------------------------------
// Tiled bf16 WMMA GEMM: 128x128 block tile, K-step 64, DOUBLE-BUFFERED
// LDS (~72KB of the WGP's 320KB). 8 waves (4x2); each wave: 2x4 16x16
// accumulators, 16 v_wmma per k-iteration. bf16 A-operands are staged
// with GLOBAL_LOAD_ASYNC_TO_LDS_B128 (ASYNCcnt); fp32 operands are
// converted to bf16 by VALU while staging. Expert mode via blockIdx.z.
// ------------------------------------------------------------------
enum { EPI_RELU2 = 0, EPI_STORE = 1, EPI_ATOMIC = 2 };
#define BK   64
#define BSTR 72   // LDS row stride (bf16 elems): even, 144B rows (16B aligned)

template <int EPI, bool ABF16>
__global__ void __launch_bounds__(256) gemm_wmma(
    const void* __restrict__ Aptr, const float* __restrict__ Bbase,
    void* __restrict__ Cptr,
    const float* __restrict__ rowScale, const int* __restrict__ rowMap,
    const int* __restrict__ mCounts, const int* __restrict__ mOffsets,
    int M, int N, int K, int lda, int ldb, int ldc, long long strideB) {
  __shared__ unsigned short As[2][128 * BSTR];
  __shared__ unsigned short Bs[2][128 * BSTR];

  int e    = blockIdx.z;
  int Mcur = mCounts ? mCounts[e] : M;
  int base = mOffsets ? mOffsets[e] : 0;
  int m0   = blockIdx.y * 128;
  if (m0 >= Mcur) return;
  int n0 = blockIdx.x * 128;
  const float* B = Bbase + (size_t)e * (size_t)strideB;

  int tid = threadIdx.x;
  int lane = tid & 31, wid = tid >> 5;
  int half = lane >> 4, mr = lane & 15;
  int mwave = (wid & 3) * 32;
  int nwave = (wid >> 2) * 64;

  v8f acc[2][4];
#pragma unroll
  for (int i = 0; i < 2; ++i)
#pragma unroll
    for (int j = 0; j < 4; ++j)
#pragma unroll
      for (int r = 0; r < 8; ++r) acc[i][j][r] = 0.f;

  // ---- tile staging ----
  auto stageA = [&](int buf, int kt) {
    if (ABF16) {
#pragma unroll
      for (int l = 0; l < 4; ++l) {       // 128x64 bf16 = 1024 x 16B
        int id = tid + l * 256;
        int rowL = id >> 3;
        int cq = (id & 7) * 8;
        int row = m0 + rowL;
        unsigned short* dst = &As[buf][rowL * BSTR + cq];
        if (row < Mcur) {
          const unsigned short* src =
              (const unsigned short*)Aptr + (size_t)(base + row) * lda + kt + cq;
#if USE_ASYNC_LDS
          async_copy16(dst, src);
#else
          *(uint4*)dst = *(const uint4*)src;
#endif
        } else {
          uint4 z = {0u, 0u, 0u, 0u};
          *(uint4*)dst = z;
        }
      }
    } else {
#pragma unroll
      for (int l = 0; l < 8; ++l) {       // 128x64 fp32 -> bf16
        int id = tid + l * 256;
        int rowL = id >> 4;
        int kq = (id & 15) * 4;
        int row = m0 + rowL;
        unsigned w0 = 0, w1 = 0;
        if (row < Mcur) {
          const float* Af = (const float*)Aptr;
          float4 f = *(const float4*)(Af + (size_t)(base + row) * lda + kt + kq);
          w0 = pack2bf(f.x, f.y);
          w1 = pack2bf(f.z, f.w);
        }
        *(unsigned*)&As[buf][rowL * BSTR + kq]     = w0;
        *(unsigned*)&As[buf][rowL * BSTR + kq + 2] = w1;
      }
    }
  };
  auto stageB = [&](int buf, int kt) {    // 64x128 fp32 -> Bs[n][k] bf16
#pragma unroll
    for (int l = 0; l < 8; ++l) {
      int id = tid + l * 256;
      int kr = id >> 5;            // 0..63
      int nq = (id & 31) * 4;      // 0..124
      float4 f = *(const float4*)(B + (size_t)(kt + kr) * ldb + n0 + nq);
      Bs[buf][(nq + 0) * BSTR + kr] = f2bf(f.x);
      Bs[buf][(nq + 1) * BSTR + kr] = f2bf(f.y);
      Bs[buf][(nq + 2) * BSTR + kr] = f2bf(f.z);
      Bs[buf][(nq + 3) * BSTR + kr] = f2bf(f.w);
    }
  };
  auto compute = [&](int buf) {
    union FR { v16bf v; unsigned u[8]; };
#pragma unroll
    for (int kk = 0; kk < BK; kk += 32) {
      FR a[2], b[4];
#pragma unroll
      for (int i = 0; i < 2; ++i) {
        int row = mwave + 16 * i + mr;
#pragma unroll
        for (int p = 0; p < 8; ++p) {
          int kb = kk + ((p & 4) ? 16 : 0) + half * 8 + (p & 3) * 2;
          a[i].u[p] = *(const unsigned*)&As[buf][row * BSTR + kb];
        }
      }
#pragma unroll
      for (int j = 0; j < 4; ++j) {
        int col = nwave + 16 * j + mr;
#pragma unroll
        for (int p = 0; p < 8; ++p) {
          int kb = kk + ((p & 4) ? 16 : 0) + half * 8 + (p & 3) * 2;
          b[j].u[p] = *(const unsigned*)&Bs[buf][col * BSTR + kb];
        }
      }
#pragma unroll
      for (int i = 0; i < 2; ++i)
#pragma unroll
        for (int j = 0; j < 4; ++j)
          acc[i][j] = __builtin_amdgcn_wmma_f32_16x16x32_bf16(
              false, a[i].v, false, b[j].v, (short)0, acc[i][j], false, false);
    }
  };

  // ---- software-pipelined main loop (1 barrier / iteration) ----
  stageA(0, 0);
  stageB(0, 0);
#if USE_ASYNC_LDS
  if (ABF16) wait_async0();
#endif
  __syncthreads();
  int c = 0;
  for (int kt = 0; kt < K; kt += BK) {
    if (kt + BK < K) { stageA(c ^ 1, kt + BK); stageB(c ^ 1, kt + BK); }
    compute(c);
#if USE_ASYNC_LDS
    if (ABF16) wait_async0();
#endif
    __syncthreads();
    c ^= 1;
  }

  // ---- epilogue ----
  float* Cf = (float*)Cptr;
  unsigned short* Cb = (unsigned short*)Cptr;
#pragma unroll
  for (int i = 0; i < 2; ++i)
#pragma unroll
    for (int j = 0; j < 4; ++j)
#pragma unroll
      for (int r = 0; r < 8; ++r) {
        int rowL = mwave + 16 * i + half * 8 + r;
        int row  = m0 + rowL;
        if (row >= Mcur) continue;
        int col = n0 + nwave + 16 * j + mr;
        float v = acc[i][j][r];
        if (EPI == EPI_RELU2) {
          v = v > 0.f ? v * v : 0.f;                 // relu^2
          if (rowScale) v *= rowScale[base + row];   // routing gate
          Cb[(size_t)(base + row) * ldc + col] = f2bf(v);
        } else if (EPI == EPI_STORE) {
          Cf[(size_t)row * ldc + col] = v;
        } else {
          int tr = rowMap[base + row];
          atomicAdd(&Cf[(size_t)tr * ldc + col], v);
        }
      }
}

// ------------------------------------------------------------------
extern "C" void kernel_launch(void* const* d_in, const int* in_sizes, int n_in,
                              void* d_out, int out_size, void* d_ws, size_t ws_size,
                              hipStream_t stream) {
  const float* x       = (const float*)d_in[0];  // [1,2048,2048]
  const float* rw      = (const float*)d_in[1];  // [16,2048]
  const float* ebias   = (const float*)d_in[2];  // [16]
  const float* w_up    = (const float*)d_in[3];  // [16,2048,1024]
  const float* w_down  = (const float*)d_in[4];  // [16,1024,2048]
  const float* sh_up   = (const float*)d_in[5];  // [2048,4096]
  const float* sh_down = (const float*)d_in[6];  // [4096,2048]
  float* out = (float*)d_out;                    // [1,2048,2048]

  char* ws = (char*)d_ws;
  unsigned short* sm  = (unsigned short*)(ws);                       // 2048x4096 bf16 (16MB)
  unsigned short* xg  = (unsigned short*)(ws + ((size_t)16 << 20));  // 4096x2048 bf16 (16MB)
  unsigned short* act = (unsigned short*)(ws + ((size_t)32 << 20));  // 4096x1024 bf16 (8MB)
  char* meta = ws + ((size_t)40 << 20);
  int*   topk_idx  = (int*)(meta);
  float* topk_w    = (float*)(meta + 16 * 1024);
  int*   token_ids = (int*)(meta + 32 * 1024);
  float* gates     = (float*)(meta + 48 * 1024);
  int*   counts    = (int*)(meta + 64 * 1024);
  int*   offsets   = (int*)(meta + 64 * 1024 + 128);
  int*   cursors   = (int*)(meta + 64 * 1024 + 256);

  zero_counts_kernel<<<1, 32, 0, stream>>>(counts);
  router_kernel<<<T_TOK / 8, 256, 0, stream>>>(x, rw, ebias, topk_idx, topk_w, counts);
  scan_kernel<<<1, 32, 0, stream>>>(counts, offsets, cursors);
  gather_kernel<<<T_TOK * 2, 256, 0, stream>>>(x, topk_idx, topk_w, cursors,
                                               token_ids, gates, xg);
  // shared expert: mid = relu2(x @ sh_up)   [2048 x 4096]
  gemm_wmma<EPI_RELU2, false><<<dim3(SHARED_INTER / 128, T_TOK / 128, 1), 256, 0, stream>>>(
      x, sh_up, sm, nullptr, nullptr, nullptr, nullptr,
      T_TOK, SHARED_INTER, HIDDEN, HIDDEN, SHARED_INTER, SHARED_INTER, 0);
  // shared expert: out = mid @ sh_down      [2048 x 2048] (plain store)
  gemm_wmma<EPI_STORE, true><<<dim3(HIDDEN / 128, T_TOK / 128, 1), 256, 0, stream>>>(
      sm, sh_down, out, nullptr, nullptr, nullptr, nullptr,
      T_TOK, HIDDEN, SHARED_INTER, SHARED_INTER, HIDDEN, HIDDEN, 0);
  // routed: act = relu2(xg @ w_up[e]) * gate   (grouped GEMM over experts)
  gemm_wmma<EPI_RELU2, true><<<dim3(MOE_INTER / 128, T_TOK / 128, N_EXPERTS), 256, 0, stream>>>(
      xg, w_up, act, gates, nullptr, counts, offsets,
      T_TOK, MOE_INTER, HIDDEN, HIDDEN, MOE_INTER, MOE_INTER,
      (long long)HIDDEN * MOE_INTER);
  // routed: out[token] += act @ w_down[e]      (atomic scatter-add)
  gemm_wmma<EPI_ATOMIC, true><<<dim3(HIDDEN / 128, T_TOK / 128, N_EXPERTS), 256, 0, stream>>>(
      act, w_down, out, nullptr, token_ids, counts, offsets,
      T_TOK, HIDDEN, MOE_INTER, MOE_INTER, HIDDEN, HIDDEN,
      (long long)MOE_INTER * HIDDEN);
}
